// PopulationReadout_58884001628734
// MI455X (gfx1250) — matrix operand order repaired and checked by reading, not compiled, not run on hardware
//
#include <hip/hip_runtime.h>

typedef _Float16 f16;
typedef __attribute__((ext_vector_type(16))) _Float16 v16h;
typedef __attribute__((ext_vector_type(8)))  float    v8f;

#define B_SZ        64
#define C_CH        64
#define HW          1296      // 36*36
#define N_UNITS     2048
#define U_PER_BLOCK 128       // 8 waves * 16 units
#define PIX_STAGE   48
#define N_STAGES    27        // 1296 / 48
#define LDS_STRIDE  72        // padded f16 row stride (144B: 16B-aligned, bank-spread)

__global__ __launch_bounds__(256)
void pop_readout_fused(const float* __restrict__ x,
                       const float* __restrict__ fw,
                       const float* __restrict__ bias,
                       const float* __restrict__ sw,
                       float* __restrict__ out)
{
    __shared__ f16 fwl[U_PER_BLOCK * LDS_STRIDE];   // 128 units x 64 ch (padded)
    __shared__ f16 xs [PIX_STAGE   * LDS_STRIDE];   // 48 px   x 64 ch (padded)

    const int t    = threadIdx.x;
    const int wave = t >> 5;
    const int lane = t & 31;
    const int half = lane >> 4;      // 0: lanes 0-15, 1: lanes 16-31
    const int ln   = lane & 15;

    const int b      = blockIdx.y;
    const int ublock = blockIdx.x * U_PER_BLOCK;
    const int ubase  = ublock + wave * 16;          // this wave's 16 units

    // ---- stage feat_weights slab (128 x 64) into LDS as f16 ----
    #pragma unroll
    for (int j = 0; j < 32; ++j) {
        int i  = t + j * 256;        // 0..8191
        int ul = i >> 6;
        int c  = i & 63;
        fwl[ul * LDS_STRIDE + c] = (f16)fw[(size_t)(ublock + ul) * C_CH + c];
    }
    __syncthreads();

    // ---- loop-invariant A fragments: f16 16x32 per ISA layout ----
    // lane row M = ln; element e<8 -> K = kbase + e + half*8
    //                  element e>=8 -> K = kbase + 16 + (e-8) + half*8
    v16h afrag[2];
    #pragma unroll
    for (int ks = 0; ks < 2; ++ks) {
        const f16* base = &fwl[(wave * 16 + ln) * LDS_STRIDE + ks * 32 + half * 8];
        #pragma unroll
        for (int e = 0; e < 8; ++e) {
            afrag[ks][e]     = base[e];
            afrag[ks][e + 8] = base[16 + e];
        }
    }

    v8f acc = {0.f, 0.f, 0.f, 0.f, 0.f, 0.f, 0.f, 0.f};
    const float* xb = x + (size_t)b * C_CH * HW;

    for (int s = 0; s < N_STAGES; ++s) {
        const int pbase = s * PIX_STAGE;
        __syncthreads();   // previous stage's reads done
        // stage x tile: 64 ch x 48 px, stored [pixel][channel] as f16
        #pragma unroll
        for (int j = 0; j < 12; ++j) {
            int i = t + j * 256;               // 0..3071
            int c = i / PIX_STAGE;
            int p = i - c * PIX_STAGE;
            xs[p * LDS_STRIDE + c] = (f16)xb[(size_t)c * HW + pbase + p];
        }
        __syncthreads();

        #pragma unroll
        for (int nt = 0; nt < 3; ++nt) {
            const int px = nt * 16;
            // B fragments: 32x16 f16; lane col N = ln, e<8 -> K = kbase + e + half*8, e>=8 -> K = kbase + 16 + (e-8) + half*8
            v16h b0, b1;
            {
                const f16* bb = &xs[(px + ln) * LDS_STRIDE + half * 8];
                #pragma unroll
                for (int e = 0; e < 8; ++e) { b0[e] = bb[e]; b0[e + 8] = bb[16 + e]; b1[e] = bb[32 + e]; b1[e + 8] = bb[48 + e]; }
            }
            v8f c8 = {0.f, 0.f, 0.f, 0.f, 0.f, 0.f, 0.f, 0.f};
            c8 = __builtin_amdgcn_wmma_f32_16x16x32_f16(false, afrag[0], false, b0,
                                                        (short)0, c8, false, false);
            c8 = __builtin_amdgcn_wmma_f32_16x16x32_f16(false, afrag[1], false, b1,
                                                        (short)0, c8, false, false);
            // fuse depthwise-spatial contraction: acc[r] += c8[r] * sw[u, pix]
            const int    pix = pbase + px + ln;
            const float* swp = sw + (size_t)(ubase + half * 8) * HW + pix;
            #pragma unroll
            for (int r = 0; r < 8; ++r)
                acc[r] += c8[r] * swp[(size_t)r * HW];
        }
    }

    // ---- reduce the 16 lanes of each half (C-tile N dimension) ----
    #pragma unroll
    for (int r = 0; r < 8; ++r) {
        float v = acc[r];
        v += __shfl_xor(v, 1, 32);
        v += __shfl_xor(v, 2, 32);
        v += __shfl_xor(v, 4, 32);
        v += __shfl_xor(v, 8, 32);
        acc[r] = v;
    }

    if (ln == 0) {
        #pragma unroll
        for (int r = 0; r < 8; ++r) {
            int u = ubase + half * 8 + r;
            out[(size_t)b * N_UNITS + u] = acc[r] + bias[u];
        }
    }
}

extern "C" void kernel_launch(void* const* d_in, const int* in_sizes, int n_in,
                              void* d_out, int out_size, void* d_ws, size_t ws_size,
                              hipStream_t stream)
{
    const float* x  = (const float*)d_in[0];   // [64,64,36,36]
    const float* fw = (const float*)d_in[1];   // [2048,64,1,1]
    const float* bs = (const float*)d_in[2];   // [2048]
    const float* sw = (const float*)d_in[3];   // [2048,36,36]
    float* out = (float*)d_out;                // [64,2048,1,1]

    dim3 grid(N_UNITS / U_PER_BLOCK, B_SZ);    // (16 unit-slabs, 64 batches)
    pop_readout_fused<<<grid, 256, 0, stream>>>(x, fw, bs, sw, out);
}
